// ShapeBasedPooling_37271726195508
// MI455X (gfx1250) — matrix operand order repaired and checked by reading, not compiled, not run on hardware
//
#include <hip/hip_runtime.h>
#include <hip/hip_bf16.h>

typedef __attribute__((ext_vector_type(16))) _Float16 v16h;
typedef __attribute__((ext_vector_type(8)))  _Float16 v8h;
typedef __attribute__((ext_vector_type(8)))  float    v8f;

#define N_PEDS     4096
#define GRID_CELLS 256      // 16x16
#define IN_DIM     512      // 256 cells * 2 (vx, vy)
#define OUT_DIM    256
#define CELL_SIDE  0.6f

// ---------------------------------------------------------------------------
// Kernel 0: convert W_emb [256, 512] f32 -> f16 (row-major, same layout).
// Rows of W are columns of B = W^T, which is exactly what the WMMA B-fragment
// loads want (contiguous K per lane).
// ---------------------------------------------------------------------------
__global__ void convert_w_f16(const float* __restrict__ W,
                              _Float16* __restrict__ Wh) {
  const int idx = blockIdx.x * blockDim.x + threadIdx.x;  // 0 .. 131071
  Wh[idx] = (_Float16)W[idx];
}

// ---------------------------------------------------------------------------
// Kernel 1: directional grid pooling. One wave (32 lanes) per pedestrian.
// Wave-private accumulator in LDS (256 cells x {vx,vy} + 256 counts = 3 KB),
// ds_add_f32 atomics resolve intra-wave bin conflicts. 8 waves/block = 24 KB
// LDS out of 320 KB per WGP. Emits the f16 A matrix [4096, 512] to workspace.
// ---------------------------------------------------------------------------
__global__ void directional_grid_kernel(const float* __restrict__ pos,
                                        const float* __restrict__ past,
                                        _Float16* __restrict__ Ah) {
  __shared__ float sval[8][2 * GRID_CELLS];   // 16 KB
  __shared__ float scnt[8][GRID_CELLS];       //  8 KB
  const int wave = threadIdx.x >> 5;
  const int lane = threadIdx.x & 31;
  const int i = blockIdx.x * 8 + wave;

  for (int c = lane; c < 2 * GRID_CELLS; c += 32) sval[wave][c] = 0.0f;
  for (int c = lane; c < GRID_CELLS;     c += 32) scnt[wave][c] = 0.0f;
  __syncthreads();

  const float pix = pos[2 * i + 0], piy = pos[2 * i + 1];
  const float vix = pix - past[2 * i + 0];
  const float viy = piy - past[2 * i + 1];

  for (int j = lane; j < N_PEDS; j += 32) {
    const float pjx = pos[2 * j + 0], pjy = pos[2 * j + 1];
    const float rx = pjx - pix;
    const float ry = pjy - piy;
    // cell index: floor(rel / cell_side + n/2), n = 16
    const int ox = (int)floorf(rx / CELL_SIDE + 8.0f);
    const int oy = (int)floorf(ry / CELL_SIDE + 8.0f);
    if (((unsigned)ox < 16u) && ((unsigned)oy < 16u) && (j != i)) {
      const float vjx = pjx - past[2 * j + 0];
      const float vjy = pjy - past[2 * j + 1];
      const int cell = ox * 16 + oy;
      atomicAdd(&sval[wave][2 * cell + 0], vjx - vix);   // ds_add_f32
      atomicAdd(&sval[wave][2 * cell + 1], vjy - viy);
      atomicAdd(&scnt[wave][cell], 1.0f);
    }
  }
  __syncthreads();

  // norm_cell (mean per occupied cell) + f32 -> f16, flat [i, cell*2 + d]
  _Float16* out = Ah + (size_t)i * IN_DIM;
  for (int c = lane; c < GRID_CELLS; c += 32) {
    const float cnt = fmaxf(scnt[wave][c], 1.0f);
    out[2 * c + 0] = (_Float16)(sval[wave][2 * c + 0] / cnt);
    out[2 * c + 1] = (_Float16)(sval[wave][2 * c + 1] / cnt);
  }
}

// ---------------------------------------------------------------------------
// Kernel 2: out = relu(A[4096,512] x W^T[512,256] + b) via
// v_wmma_f32_16x16x32_f16. One wave per 16x16 output tile; block = 8 waves
// covering 16 rows x 128 cols; grid = (256 row-tiles, 2 col-groups).
// Fragment loads use contiguous K per lane half (same order for A and B):
//   A lane l: row = l%16; halves [0..15] = K 16g..16g+15
//   B lane l: col = l%16; halves [0..15] = K 16g..16g+15  (row n of W)
//   C/D VGPR r: row = r + 8g, col = l%16
// ---------------------------------------------------------------------------
__global__ void emb_gemm_wmma(const _Float16* __restrict__ Ah,
                              const _Float16* __restrict__ Wh,
                              const float* __restrict__ bias,
                              float* __restrict__ out) {
  const int wave = threadIdx.x >> 5;
  const int lane = threadIdx.x & 31;
  const int g    = lane >> 4;            // which half of the wave
  const int ln   = lane & 15;
  const int m0 = blockIdx.x * 16;
  const int n0 = blockIdx.y * 128 + wave * 16;

  const _Float16* Arow = Ah + (size_t)(m0 + ln) * IN_DIM;
  const _Float16* Brow = Wh + (size_t)(n0 + ln) * IN_DIM;  // W row n = col n of W^T

  v8f c = {};
#pragma unroll
  for (int kk = 0; kk < IN_DIM / 32; ++kk) {
    const int k0 = kk * 32;
    union { v16h v; v8h h[2]; } a;
    a.v = *(const v16h*)(Arow + k0 + 16 * g);         // K = k0+16g .. +15
    const v16h b = *(const v16h*)(Brow + k0 + 16 * g); // K = k0+16g .. +15
    c = __builtin_amdgcn_wmma_f32_16x16x32_f16(
        /*neg_a=*/false, a.v, /*neg_b=*/false, b,
        /*c_mod=*/(short)0, c, /*reuse_a=*/false, /*reuse_b=*/false);
  }

  const float bn = bias[n0 + ln];
#pragma unroll
  for (int r = 0; r < 8; ++r) {
    const int row = m0 + r + 8 * g;
    const float v = c[r] + bn;
    out[(size_t)row * OUT_DIM + (n0 + ln)] = v > 0.0f ? v : 0.0f;
  }
}

// ---------------------------------------------------------------------------
// Launch. Inputs (setup_inputs order): h [4096*128] (UNUSED by reference),
// positions [4096*2], past_positions [4096*2], W_emb [256*512], b_emb [256].
// Output: [4096, 256] f32. Workspace: A_f16 (4 MB) + W_f16 (256 KB).
// ---------------------------------------------------------------------------
extern "C" void kernel_launch(void* const* d_in, const int* in_sizes, int n_in,
                              void* d_out, int out_size, void* d_ws, size_t ws_size,
                              hipStream_t stream) {
  (void)in_sizes; (void)n_in; (void)out_size; (void)ws_size;
  const float* pos  = (const float*)d_in[1];
  const float* past = (const float*)d_in[2];
  const float* W    = (const float*)d_in[3];
  const float* bias = (const float*)d_in[4];
  float* out = (float*)d_out;

  _Float16* Ah = (_Float16*)d_ws;                                      // 4 MB
  _Float16* Wh = (_Float16*)((char*)d_ws + (size_t)N_PEDS * IN_DIM * sizeof(_Float16));

  convert_w_f16<<<(OUT_DIM * IN_DIM) / 256, 256, 0, stream>>>(W, Wh);
  directional_grid_kernel<<<N_PEDS / 8, 256, 0, stream>>>(pos, past, Ah);
  dim3 grid(N_PEDS / 16, OUT_DIM / 128);
  emb_gemm_wmma<<<grid, 256, 0, stream>>>(Ah, Wh, bias, out);
}
